// GAT_23270132810343
// MI455X (gfx1250) — matrix operand order repaired and checked by reading, not compiled, not run on hardware
//
#include <hip/hip_runtime.h>

#define NN 2048
#define DD 256
#define HH 8
#define DH 32
#define BB 2
#define NDEPTH 2

typedef __attribute__((ext_vector_type(16))) __bf16 v16bf;
typedef __attribute__((ext_vector_type(8)))  float  v8f;

union Frag { v16bf v; unsigned int u[8]; };
union Acc  { v8f v; float f[8]; };

// round-to-nearest-even f32 -> bf16 bits
__device__ __forceinline__ unsigned int f2bf1(float f) {
    unsigned int u = __float_as_uint(f);
    return (u + 0x7FFFu + ((u >> 16) & 1u)) >> 16;
}
__device__ __forceinline__ unsigned int packbf2(float lo, float hi) {
    return f2bf1(lo) | (f2bf1(hi) << 16);
}
__device__ __forceinline__ float bflo(unsigned int w) { return __uint_as_float(w << 16); }
__device__ __forceinline__ float bfhi(unsigned int w) { return __uint_as_float(w & 0xFFFF0000u); }

// CDNA5 async global->LDS copy, 16 bytes per lane (tracked by ASYNCcnt)
__device__ __forceinline__ void async_ld16(unsigned ldsoff, const void* g) {
    asm volatile("global_load_async_to_lds_b128 %0, %1, off"
                 : : "v"(ldsoff), "v"(g) : "memory");
}
__device__ __forceinline__ void wait_async_le2() {
    asm volatile("s_wait_asynccnt 0x2" ::: "memory");
}
__device__ __forceinline__ void wait_async0() {
    asm volatile("s_wait_asynccnt 0x0" ::: "memory");
}

// ---------------------------------------------------------------------------
// f32 -> bf16 bulk convert (2048 elems / 256-thread block)
// ---------------------------------------------------------------------------
__global__ __launch_bounds__(256) void cvt_bf16(const float* __restrict__ x,
                                                unsigned short* __restrict__ y)
{
    long long e0 = ((long long)blockIdx.x * 256 + threadIdx.x) * 8;
    const float4* p = (const float4*)(x + e0);
    float4 a = p[0], b = p[1];
    uint4 o;
    o.x = packbf2(a.x, a.y); o.y = packbf2(a.z, a.w);
    o.z = packbf2(b.x, b.y); o.w = packbf2(b.z, b.w);
    *(uint4*)(y + e0) = o;
}

// ---------------------------------------------------------------------------
// Y[M,N] = A[M,K](bf16) * B(bf16)   (TRANSB: B is [N,K] -> Y = A*B^T;
//                                    else  : B is [K,N] -> Y = A*B)
// Double-buffered LDS tiles, DMA'd with global_load_async_to_lds_b128 and
// software-pipelined against the WMMA stream (wait asynccnt<=2 keeps the
// next tile's copies in flight). C stored f32 and/or bf16. Batched via z.
// ---------------------------------------------------------------------------
template<int TRANSB>
__global__ __launch_bounds__(256) void gemm_bf16(
    const unsigned short* __restrict__ A, const unsigned short* __restrict__ Bm,
    float* __restrict__ C, unsigned short* __restrict__ Cb,
    int N, int K, int ldb,
    long long sA, long long sB, long long sC)
{
    __shared__ unsigned short As[2][64 * 40];    // [m][k] rows padded to 80B (16B-aligned)
    __shared__ unsigned short Bsh[2][64 * 40];   // TRANSB: [n][40]; else: [k][72]

    const int tid  = threadIdx.x;
    const int lane = tid & 31;
    const int w    = tid >> 5;
    const int wm   = w >> 1;     // 0..3
    const int wn   = w & 1;      // 0..1
    const int hi16 = lane >> 4;
    const int l15  = lane & 15;

    const long long bz = blockIdx.z;
    A += bz * sA;  Bm += bz * sB;

    const int m0 = blockIdx.y * 64;
    const int n0 = blockIdx.x * 64;

    // per-thread async-copy coordinates (one b128 each per tile)
    const int ar = tid >> 2, aseg = tid & 3;     // A / B-TRANSB: 64 rows x 4 segs
    const int bk = tid >> 3, bseg = tid & 7;     // B-normal:     32 rows x 8 segs

    auto issue = [&](int k0, int buf) {
        async_ld16((unsigned)(size_t)&As[buf][ar * 40 + aseg * 8],
                   A + (long long)(m0 + ar) * K + k0 + aseg * 8);
        if (TRANSB) {
            async_ld16((unsigned)(size_t)&Bsh[buf][ar * 40 + aseg * 8],
                       Bm + (long long)(n0 + ar) * ldb + k0 + aseg * 8);
        } else {
            async_ld16((unsigned)(size_t)&Bsh[buf][bk * 72 + bseg * 8],
                       Bm + (long long)(k0 + bk) * ldb + n0 + bseg * 8);
        }
    };

    Acc c0, c1;
#pragma unroll
    for (int i = 0; i < 8; i++) { c0.f[i] = 0.f; c1.f[i] = 0.f; }

    const int niter = K >> 5;
    issue(0, 0);
    for (int it = 0; it < niter; it++) {
        const int k0  = it << 5;
        const int cur = it & 1;
        if (it + 1 < niter) {
            issue(k0 + 32, cur ^ 1);                      // overlap next tile's DMA
            if (k0 + 64 < K)
                __builtin_prefetch(A + (long long)(m0 + ar) * K + k0 + 64 + aseg * 8, 0, 3);
            wait_async_le2();                             // tile `it` landed; next in flight
        } else {
            wait_async0();
        }
        __syncthreads();

        Frag a, b0, b1;
#pragma unroll
        for (int j = 0; j < 8; j++) {
            int kk = hi16 * 16 + 2 * j;
            a.u[j] = *(const unsigned int*)&As[cur][(16 * wm + l15) * 40 + kk];
        }
        if (TRANSB) {
#pragma unroll
            for (int j = 0; j < 8; j++) {
                int kk = hi16 * 16 + 2 * j;
                b0.u[j] = *(const unsigned int*)&Bsh[cur][(32 * wn + l15) * 40 + kk];
                b1.u[j] = *(const unsigned int*)&Bsh[cur][(32 * wn + 16 + l15) * 40 + kk];
            }
        } else {
#pragma unroll
            for (int j = 0; j < 8; j++) {
                int kr = hi16 * 16 + 2 * j;
                int c  = 32 * wn + l15;
                b0.u[j] = (unsigned)Bsh[cur][kr * 72 + c]
                        | ((unsigned)Bsh[cur][(kr + 1) * 72 + c] << 16);
                b1.u[j] = (unsigned)Bsh[cur][kr * 72 + c + 16]
                        | ((unsigned)Bsh[cur][(kr + 1) * 72 + c + 16] << 16);
            }
        }
        c0.v = __builtin_amdgcn_wmma_f32_16x16x32_bf16(false, a.v, false, b0.v, (short)0, c0.v, false, false);
        c1.v = __builtin_amdgcn_wmma_f32_16x16x32_bf16(false, a.v, false, b1.v, (short)0, c1.v, false, false);
        __syncthreads();                                  // readers done before buffer reuse
    }

#pragma unroll
    for (int r = 0; r < 8; r++) {
        long long row = m0 + 16 * wm + r + hi16 * 8;
        int col = n0 + 32 * wn + l15;
        if (C) {
            float* cp = C + bz * sC + row * N + col;
            cp[0]  = c0.f[r];
            cp[16] = c1.f[r];
        }
        if (Cb) {
            unsigned short* cp = Cb + bz * sC + row * N + col;
            cp[0]  = (unsigned short)f2bf1(c0.f[r]);
            cp[16] = (unsigned short)f2bf1(c1.f[r]);
        }
    }
}

// ---------------------------------------------------------------------------
// Fused masked attention on bf16 q/k/v. One WG per (batch, 16-row q tile);
// wave w = head w.
// Pass 1: lane-wise accumulation of exp(s) (masked -> exp(-1e9)=0); one
//         4-stage butterfly per row at the end (no per-tile shuffles).
// Pass 2: p = exp(s)/l, head-average into agg (bf16, no atomics),
//         o += p*v per head (bf16 out). dh=32 == one bf16 WMMA K-step.
// ---------------------------------------------------------------------------
__global__ __launch_bounds__(256) void gat_attn(
    const unsigned short* __restrict__ q, const unsigned short* __restrict__ k,
    const unsigned short* __restrict__ v, const float* __restrict__ adj,
    unsigned short* __restrict__ o, unsigned short* __restrict__ agg)
{
    __shared__ unsigned int   adjbits[16][64];   // 16 rows x 2048 bits
    __shared__ unsigned short plds[HH][16][32];  // per-head p tile (bf16)

    const int tid  = threadIdx.x;
    const int lane = tid & 31;
    const int h    = tid >> 5;
    const int b    = blockIdx.y;
    const int q0   = blockIdx.x * 16;
    const int hi16 = lane >> 4;
    const int l15  = lane & 15;
    const float scale = 0.17677669529663687f;    // 1/sqrt(32)

    // phase 0: adjacency bitmask (float4-vectorized)
#pragma unroll
    for (int i = 0; i < 4; i++) {
        int word = tid + i * 256;
        int r = word >> 6, wd = word & 63;
        const float4* arow = (const float4*)(adj + (long long)(q0 + r) * NN + wd * 32);
        unsigned int mword = 0u;
#pragma unroll
        for (int g = 0; g < 8; g++) {
            float4 a4 = arow[g];
            mword |= (a4.x > 0.f ? 1u : 0u) << (4 * g);
            mword |= (a4.y > 0.f ? 1u : 0u) << (4 * g + 1);
            mword |= (a4.z > 0.f ? 1u : 0u) << (4 * g + 2);
            mword |= (a4.w > 0.f ? 1u : 0u) << (4 * g + 3);
        }
        adjbits[r][wd] = mword;
    }
    __syncthreads();

    // q fragment: direct packed-bf16 loads
    const unsigned short* qb = q + ((long long)b * NN + q0) * DD + h * DH;
    Frag qa;
#pragma unroll
    for (int j = 0; j < 8; j++) {
        int kk = hi16 * 16 + 2 * j;
        qa.u[j] = *(const unsigned int*)(qb + (long long)l15 * DD + kk);
    }

    const unsigned short* kb  = k + (long long)b * NN * DD + h * DH;
    const unsigned short* vbp = v + (long long)b * NN * DD + h * DH;

    // ---- pass 1: lane-local partial row sums of exp(s) ----
    float rsum[8];
#pragma unroll
    for (int r = 0; r < 8; r++) rsum[r] = 0.f;

    for (int kt = 0; kt < NN; kt += 32) {
#pragma unroll
        for (int sub = 0; sub < 2; sub++) {
            int kk0 = kt + sub * 16;
            Frag bf;
#pragma unroll
            for (int j = 0; j < 8; j++) {
                int dd = hi16 * 16 + 2 * j;
                bf.u[j] = *(const unsigned int*)(kb + (long long)(kk0 + l15) * DD + dd);
            }
            Acc s;
#pragma unroll
            for (int i = 0; i < 8; i++) s.f[i] = 0.f;
            s.v = __builtin_amdgcn_wmma_f32_16x16x32_bf16(false, qa.v, false, bf.v, (short)0, s.v, false, false);

            int word   = kk0 >> 5;
            int bitpos = (kk0 & 16) + l15;
#pragma unroll
            for (int r = 0; r < 8; r++) {
                int row = r + hi16 * 8;
                unsigned int bits = adjbits[row][word];
                float sv = ((bits >> bitpos) & 1u) ? s.f[r] * scale : -1.0e9f;
                rsum[r] += __expf(sv);
            }
        }
    }
    // one butterfly per row across the 16-lane group
    float invl[8];
#pragma unroll
    for (int r = 0; r < 8; r++) {
        float e = rsum[r];
        e += __shfl_xor(e, 1, 32);
        e += __shfl_xor(e, 2, 32);
        e += __shfl_xor(e, 4, 32);
        e += __shfl_xor(e, 8, 32);
        invl[r] = 1.f / e;
    }

    Acc o0, o1;
#pragma unroll
    for (int i = 0; i < 8; i++) { o0.f[i] = 0.f; o1.f[i] = 0.f; }

    // ---- pass 2: p, head-averaged agg (bf16), o += p*v ----
    for (int kt = 0; kt < NN; kt += 32) {
#pragma unroll
        for (int sub = 0; sub < 2; sub++) {
            int kk0 = kt + sub * 16;
            Frag bf;
#pragma unroll
            for (int j = 0; j < 8; j++) {
                int dd = hi16 * 16 + 2 * j;
                bf.u[j] = *(const unsigned int*)(kb + (long long)(kk0 + l15) * DD + dd);
            }
            Acc s;
#pragma unroll
            for (int i = 0; i < 8; i++) s.f[i] = 0.f;
            s.v = __builtin_amdgcn_wmma_f32_16x16x32_bf16(false, qa.v, false, bf.v, (short)0, s.v, false, false);
            int word   = kk0 >> 5;
            int bitpos = (kk0 & 16) + l15;
#pragma unroll
            for (int r = 0; r < 8; r++) {
                int row = r + hi16 * 8;
                unsigned int bits = adjbits[row][word];
                float sv = ((bits >> bitpos) & 1u) ? s.f[r] * scale : -1.0e9f;
                float p = __expf(sv) * invl[r];
                plds[h][row][sub * 16 + l15] = (unsigned short)f2bf1(p);
            }
        }
        __syncthreads();

        // cooperative head average -> agg[b, q0+row, kt+col] (bf16 pair store)
        {
            int r  = tid >> 4;
            int cp = (tid & 15) * 2;
            float s0 = 0.f, s1 = 0.f;
#pragma unroll
            for (int hh = 0; hh < HH; hh++) {
                unsigned int wv = *(const unsigned int*)&plds[hh][r][cp];
                s0 += bflo(wv);
                s1 += bfhi(wv);
            }
            unsigned short* ap = agg + ((long long)b * NN + q0 + r) * NN + kt + cp;
            *(unsigned int*)ap = packbf2(s0 * 0.125f, s1 * 0.125f);
        }

        // o accumulation: A = p (own head), B = v tile (32k x 16d) x2
        Frag pa;
#pragma unroll
        for (int j = 0; j < 8; j++) {
            int kk = hi16 * 16 + 2 * j;
            pa.u[j] = *(const unsigned int*)&plds[h][l15][kk];
        }
        Frag vb0, vb1;
#pragma unroll
        for (int j = 0; j < 8; j++) {
            int kr = hi16 * 16 + 2 * j;
            const unsigned short* pv = vbp + (long long)(kt + kr) * DD + l15;
            vb0.u[j] = (unsigned)pv[0]  | ((unsigned)pv[DD] << 16);
            vb1.u[j] = (unsigned)pv[16] | ((unsigned)pv[DD + 16] << 16);
        }
        o0.v = __builtin_amdgcn_wmma_f32_16x16x32_bf16(false, pa.v, false, vb0.v, (short)0, o0.v, false, false);
        o1.v = __builtin_amdgcn_wmma_f32_16x16x32_bf16(false, pa.v, false, vb1.v, (short)0, o1.v, false, false);
        __syncthreads();
    }

    unsigned short* ob = o + ((long long)b * NN + q0) * DD + h * DH;
#pragma unroll
    for (int r = 0; r < 8; r++) {
        int row = r + hi16 * 8;
        ob[(long long)row * DD + l15]      = (unsigned short)f2bf1(o0.f[r]);
        ob[(long long)row * DD + l15 + 16] = (unsigned short)f2bf1(o1.f[r]);
    }
}

// ---------------------------------------------------------------------------
// out = LayerNorm(x + o) * gamma + beta  (one wave per 256-wide row)
// Optionally also writes a bf16 copy for downstream WMMA consumers.
// ---------------------------------------------------------------------------
__global__ __launch_bounds__(256) void add_ln(
    const float* __restrict__ x, const float* __restrict__ o,
    const float* __restrict__ gamma, const float* __restrict__ beta,
    float* __restrict__ out, unsigned short* __restrict__ out16)
{
    const int lane = threadIdx.x & 31;
    const int w    = threadIdx.x >> 5;
    const long long row = (long long)blockIdx.x * 8 + w;
    const float* xr  = x + row * DD;
    const float* orw = o + row * DD;
    float vals[8];
    float s = 0.f, ss = 0.f;
#pragma unroll
    for (int j = 0; j < 8; j++) {
        int idx = lane + j * 32;
        float t = xr[idx] + orw[idx];
        vals[j] = t; s += t; ss += t * t;
    }
#pragma unroll
    for (int off = 16; off >= 1; off >>= 1) {
        s  += __shfl_xor(s, off, 32);
        ss += __shfl_xor(ss, off, 32);
    }
    float mean = s * (1.f / DD);
    float var  = ss * (1.f / DD) - mean * mean;
    float rs   = rsqrtf(var + 1e-5f);
    float* outr = out + row * DD;
#pragma unroll
    for (int j = 0; j < 8; j++) {
        int idx = lane + j * 32;
        float y = (vals[j] - mean) * rs * gamma[idx] + beta[idx];
        outr[idx] = y;
        if (out16) out16[row * DD + idx] = (unsigned short)f2bf1(y);
    }
}

extern "C" void kernel_launch(void* const* d_in, const int* in_sizes, int n_in,
                              void* d_out, int out_size, void* d_ws, size_t ws_size,
                              hipStream_t stream)
{
    (void)in_sizes; (void)n_in; (void)out_size; (void)ws_size;

    const float* x_in  = (const float*)d_in[0];
    const float* adj   = (const float*)d_in[1];
    const float* Wf0   = (const float*)d_in[2];
    const float* Wq    = (const float*)d_in[3];
    const float* Wk    = (const float*)d_in[4];
    const float* Wv    = (const float*)d_in[5];
    const float* Wo    = (const float*)d_in[6];
    const float* gamma = (const float*)d_in[7];
    const float* beta  = (const float*)d_in[8];

    float* out_x   = (float*)d_out;
    float* out_att = out_x + (long long)BB * NN * DD;

    const long long SXD = (long long)BB * NN * DD;   // 1,048,576
    const long long SNN = (long long)BB * NN * NN;   // 8,388,608
    const long long SW  = (long long)DD * DD;        // 65,536

    char* w8 = (char*)d_ws;
    float* xb  = (float*)w8;  w8 += SXD * 4;         // features f32 (residual path)
    float* pbf = (float*)w8;  w8 += SXD * 4;         // Wo-projected output f32
    unsigned short* xin16 = (unsigned short*)w8; w8 += SXD * 2;
    unsigned short* x16   = (unsigned short*)w8; w8 += SXD * 2;
    unsigned short* q16   = (unsigned short*)w8; w8 += SXD * 2;
    unsigned short* k16   = (unsigned short*)w8; w8 += SXD * 2;
    unsigned short* v16   = (unsigned short*)w8; w8 += SXD * 2;
    unsigned short* o16   = (unsigned short*)w8; w8 += SXD * 2;
    unsigned short* att0  = (unsigned short*)w8; w8 += SNN * 2;
    unsigned short* agg1  = (unsigned short*)w8; w8 += SNN * 2;
    unsigned short* wf16  = (unsigned short*)w8; w8 += SW * 2;
    unsigned short* wq16  = (unsigned short*)w8; w8 += NDEPTH * SW * 2;
    unsigned short* wk16  = (unsigned short*)w8; w8 += NDEPTH * SW * 2;
    unsigned short* wv16  = (unsigned short*)w8; w8 += NDEPTH * SW * 2;
    unsigned short* wo16  = (unsigned short*)w8; w8 += NDEPTH * SW * 2;

    dim3 blk(256);
    dim3 gproj(DD / 64, (BB * NN) / 64, 1);

    // one-time bf16 conversions
    cvt_bf16<<<dim3(SXD / 2048), blk, 0, stream>>>(x_in, xin16);
    cvt_bf16<<<dim3(SW / 2048), blk, 0, stream>>>(Wf0, wf16);
    cvt_bf16<<<dim3(NDEPTH * SW / 2048), blk, 0, stream>>>(Wq, wq16);
    cvt_bf16<<<dim3(NDEPTH * SW / 2048), blk, 0, stream>>>(Wk, wk16);
    cvt_bf16<<<dim3(NDEPTH * SW / 2048), blk, 0, stream>>>(Wv, wv16);
    cvt_bf16<<<dim3(NDEPTH * SW / 2048), blk, 0, stream>>>(Wo, wo16);

    // x = x_in @ Wf0^T  (f32 for residual + bf16 for projections)
    gemm_bf16<1><<<gproj, blk, 0, stream>>>(xin16, wf16, xb, x16,
                                            DD, DD, DD, 0LL, 0LL, 0LL);

    for (int layer = 0; layer < NDEPTH; layer++) {
        const unsigned short* wq = wq16 + (long long)layer * SW;
        const unsigned short* wk = wk16 + (long long)layer * SW;
        const unsigned short* wv = wv16 + (long long)layer * SW;
        const unsigned short* wo = wo16 + (long long)layer * SW;

        gemm_bf16<1><<<gproj, blk, 0, stream>>>(x16, wq, (float*)nullptr, q16,
                                                DD, DD, DD, 0LL, 0LL, 0LL);
        gemm_bf16<1><<<gproj, blk, 0, stream>>>(x16, wk, (float*)nullptr, k16,
                                                DD, DD, DD, 0LL, 0LL, 0LL);
        gemm_bf16<1><<<gproj, blk, 0, stream>>>(x16, wv, (float*)nullptr, v16,
                                                DD, DD, DD, 0LL, 0LL, 0LL);

        unsigned short* agg = (layer == 0) ? att0 : agg1;
        gat_attn<<<dim3(NN / 16, BB), blk, 0, stream>>>(q16, k16, v16, adj, o16, agg);

        gemm_bf16<1><<<gproj, blk, 0, stream>>>(o16, wo, pbf, (unsigned short*)nullptr,
                                                DD, DD, DD, 0LL, 0LL, 0LL);

        float* xdst = (layer == NDEPTH - 1) ? out_x : xb;
        unsigned short* x16dst = (layer == NDEPTH - 1) ? (unsigned short*)nullptr : x16;
        add_ln<<<dim3((BB * NN) / 8), blk, 0, stream>>>(
            xb, pbf, gamma + layer * DD, beta + layer * DD, xdst, x16dst);
    }

    // final atten = agg1 @ att0 (batched over B), f32 to d_out
    gemm_bf16<0><<<dim3(NN / 64, NN / 64, BB), blk, 0, stream>>>(
        agg1, att0, out_att, (unsigned short*)nullptr, NN, NN, NN,
        (long long)NN * NN, (long long)NN * NN, (long long)NN * NN);
}
